// Conv3DCaps_52862457479580
// MI455X (gfx1250) — matrix operand order repaired and checked by reading, not compiled, not run on hardware
//
#include <hip/hip_runtime.h>

#define CI 32
#define NI 8
#define CO 32
#define NO 8
#define ITERATIONS 3

// CDNA5 async global->LDS staging (GLOBAL_LOAD_ASYNC_TO_LDS_B128, ASYNCcnt).
// Inline asm per CDNA5 ISA; GVS mode: lds_vgpr, voffset, saddr.
#define USE_ASYNC_LDS 1

typedef __attribute__((ext_vector_type(16))) __bf16 v16bf;
typedef __attribute__((ext_vector_type(8)))  float  v8f;
typedef unsigned short ushort_t;
typedef unsigned int   uint_t;

union FragB {
    v16bf v;
    uint4 q[2];
};

__device__ __forceinline__ ushort_t f2bf(float f) {
    unsigned int u = __float_as_uint(f);
    unsigned int r = u + 0x7FFFu + ((u >> 16) & 1u);   // RNE truncate to bf16
    return (ushort_t)(r >> 16);
}
__device__ __forceinline__ float bf2f(ushort_t h) {
    return __uint_as_float(((unsigned int)h) << 16);
}
__device__ __forceinline__ void unpack8(uint4 q, float* f) {
    f[0] = bf2f((ushort_t)(q.x & 0xffff)); f[1] = bf2f((ushort_t)(q.x >> 16));
    f[2] = bf2f((ushort_t)(q.y & 0xffff)); f[3] = bf2f((ushort_t)(q.y >> 16));
    f[4] = bf2f((ushort_t)(q.z & 0xffff)); f[5] = bf2f((ushort_t)(q.z >> 16));
    f[6] = bf2f((ushort_t)(q.w & 0xffff)); f[7] = bf2f((ushort_t)(q.w >> 16));
}

// ---------------------------------------------------------------------------
// Kernel 0: swizzle conv weights (256 x 72 fp32) into bf16 A-fragment order.
// A-frag (16x32 bf16, ISA 7.12.2): lanes 0-15 row M=lane, halves t<8 -> K=t,
// t>=8 -> K=16+(t-8); lanes 16-31 same rows, K offset +8.
// Layout: wfrag[mt][s][lane][t], K_global = s*32 + K_in32, zero-pad K>=72.
// ---------------------------------------------------------------------------
__global__ __launch_bounds__(256) void prep_wfrag(const float* __restrict__ cw,
                                                  ushort_t* __restrict__ wfrag) {
    int idx = blockIdx.x * 256 + threadIdx.x;          // 16*3*32*16 = 24576
    if (idx >= 16 * 3 * 32 * 16) return;
    int t    = idx & 15;
    int lane = (idx >> 4) & 31;
    int s    = (idx >> 9) % 3;
    int mt   = idx / 1536;
    int lo   = (lane < 16) ? 0 : 8;
    int Kin  = (t < 8) ? (lo + t) : (16 + lo + (t - 8));
    int K    = s * 32 + Kin;
    int M    = mt * 16 + (lane & 15);
    ushort_t val = 0;
    if (K < NI * 9) val = f2bf(cw[M * (NI * 9) + K]);  // (n,kh,kw) row-major == K
    wfrag[idx] = val;
}

// ---------------------------------------------------------------------------
// Kernel 1: batched conv-as-GEMM with v_wmma_f32_16x16x32_bf16.
// One block per (b,ci): M=256 out-ch, N=256 spatial (n_tile = h, n = w), K=96.
// Writes votes bf16, layout [b][ci][h][w][o], o = co*8+no.
// ---------------------------------------------------------------------------
__global__ __launch_bounds__(256) void conv_votes(const float* __restrict__ x,
                                                  const ushort_t* __restrict__ wfrag,
                                                  const float* __restrict__ conv_b,
                                                  const int* __restrict__ qb,
                                                  ushort_t* __restrict__ votes) {
    __shared__ ushort_t xs[NI * 256];        // x slab, bf16, 4KB
    __shared__ uint4    bfq[48 * 32 * 2];    // 48 B-frags (h=16 x s=3), 48KB

    const int blk  = blockIdx.x;             // b*CI + ci
    const int tid  = threadIdx.x;
    const int lane = tid & 31;
    const int wv   = tid >> 5;
    const int w    = lane & 15;

    const float* xg = x + (size_t)blk * (NI * 256);

#if USE_ASYNC_LDS
    // Stage fp32 slab into LDS via async DMA (aliases bfq, used before bfq).
    {
        float* xsf = (float*)bfq;            // 8KB staging inside 48KB region
        unsigned ldsbase = (unsigned)(unsigned long long)(void*)xsf;
        unsigned long long gbase = (unsigned long long)xg;
        for (int i = tid; i < 512; i += 256) {           // 512 x 16B = 8KB
            unsigned laddr = ldsbase + (unsigned)i * 16u;
            unsigned goff  = (unsigned)i * 16u;
            asm volatile("global_load_async_to_lds_b128 %0, %1, %2 offset:0"
                         :: "v"(laddr), "v"(goff), "s"(gbase) : "memory");
        }
        asm volatile("s_wait_asynccnt 0" ::: "memory");
        __syncthreads();
        for (int i = tid; i < NI * 256; i += 256) xs[i] = f2bf(xsf[i]);
    }
#else
    for (int i = tid; i < NI * 256; i += 256) xs[i] = f2bf(xg[i]);
#endif
    __syncthreads();

    // build B fragments: frag f = h*3 + s; lane n = w; halves t: K = s*32 + (lane<16?0:8) + (t<8 ? t : t+8)
    for (int p = 0; p < 6; ++p) {
        int f = p * 8 + wv;
        int h = f / 3, s = f - h * 3;
        uint_t pk[8];
#pragma unroll
        for (int tt = 0; tt < 8; ++tt) {
            ushort_t v01[2];
#pragma unroll
            for (int half = 0; half < 2; ++half) {
                int t = 2 * tt + half;
                int K = s * 32 + ((lane < 16) ? 0 : 8) + ((t < 8) ? t : (t + 8));
                ushort_t val = 0;
                if (K < NI * 9) {
                    int nidx = K / 9, r = K - nidx * 9;
                    int kh = r / 3, kw = r - kh * 3;
                    int hh = h + kh - 1, ww = w + kw - 1;
                    if (hh >= 0 && hh < 16 && ww >= 0 && ww < 16)
                        val = xs[nidx * 256 + hh * 16 + ww];
                }
                v01[half] = val;
            }
            pk[tt] = (uint_t)v01[0] | ((uint_t)v01[1] << 16);
        }
        int base = (f * 32 + lane) * 2;
        bfq[base + 0] = make_uint4(pk[0], pk[1], pk[2], pk[3]);
        bfq[base + 1] = make_uint4(pk[4], pk[5], pk[6], pk[7]);
    }
    __syncthreads();

    const float qs    = (float)(1 << qb[0]);
    const float invqs = 1.0f / qs;
    const int   obase_off = (lane < 16) ? 0 : 8;

    for (int mi = 0; mi < 2; ++mi) {
        int mt = wv + mi * 8;                 // m-tile 0..15
        FragB a[3];
#pragma unroll
        for (int s = 0; s < 3; ++s) {
            const uint4* ap = (const uint4*)(wfrag + (((mt * 3 + s) * 32) + lane) * 16);
            a[s].q[0] = ap[0];
            a[s].q[1] = ap[1];
        }
        int obase = mt * 16 + obase_off;      // 8 contiguous out channels per lane
        float bias8[8];
#pragma unroll
        for (int r = 0; r < 8; ++r) bias8[r] = conv_b[obase + r];

        for (int h = 0; h < 16; ++h) {
            v8f acc = {0.f, 0.f, 0.f, 0.f, 0.f, 0.f, 0.f, 0.f};
#pragma unroll
            for (int s = 0; s < 3; ++s) {
                FragB bb;
                int bbase = ((h * 3 + s) * 32 + lane) * 2;
                bb.q[0] = bfq[bbase + 0];
                bb.q[1] = bfq[bbase + 1];
                acc = __builtin_amdgcn_wmma_f32_16x16x32_bf16(
                    false, a[s].v, false, bb.v, (short)0, acc, false, false);
            }
            // epilogue: + conv_b, quant(2^-qb), pack 8 contiguous bf16, one 16B store
            uint4 o;
            float v0, v1;
            v0 = rintf((acc[0] + bias8[0]) * qs) * invqs;
            v1 = rintf((acc[1] + bias8[1]) * qs) * invqs;
            o.x = (uint_t)f2bf(v0) | ((uint_t)f2bf(v1) << 16);
            v0 = rintf((acc[2] + bias8[2]) * qs) * invqs;
            v1 = rintf((acc[3] + bias8[3]) * qs) * invqs;
            o.y = (uint_t)f2bf(v0) | ((uint_t)f2bf(v1) << 16);
            v0 = rintf((acc[4] + bias8[4]) * qs) * invqs;
            v1 = rintf((acc[5] + bias8[5]) * qs) * invqs;
            o.z = (uint_t)f2bf(v0) | ((uint_t)f2bf(v1) << 16);
            v0 = rintf((acc[6] + bias8[6]) * qs) * invqs;
            v1 = rintf((acc[7] + bias8[7]) * qs) * invqs;
            o.w = (uint_t)f2bf(v0) | ((uint_t)f2bf(v1) << 16);
            size_t vbase = ((((size_t)blk * 16 + h) * 16 + w) * 256) + obase;
            *(uint4*)(votes + vbase) = o;
        }
    }
}

// ---------------------------------------------------------------------------
__global__ __launch_bounds__(256) void zero_f32(float* __restrict__ p, int n) {
    int i = blockIdx.x * 256 + threadIdx.x;
    if (i < n) p[i] = 0.0f;
}

// softmax over (co,h,w)=8192 per (b,ci); route = quant(softmax, qb) stored bf16
__global__ __launch_bounds__(256) void softmax_route(const float* __restrict__ logits,
                                                     ushort_t* __restrict__ route,
                                                     const int* __restrict__ qb) {
    __shared__ float red[256];
    const int tid = threadIdx.x;
    const float* lp = logits + (size_t)blockIdx.x * 8192;
    float e[32];
    float m = -3.0e38f;
#pragma unroll
    for (int i = 0; i < 32; ++i) { e[i] = lp[tid + i * 256]; m = fmaxf(m, e[i]); }
    red[tid] = m; __syncthreads();
    for (int off = 128; off > 0; off >>= 1) {
        if (tid < off) red[tid] = fmaxf(red[tid], red[tid + off]);
        __syncthreads();
    }
    m = red[0]; __syncthreads();
    float s = 0.0f;
#pragma unroll
    for (int i = 0; i < 32; ++i) { e[i] = __expf(e[i] - m); s += e[i]; }
    red[tid] = s; __syncthreads();
    for (int off = 128; off > 0; off >>= 1) {
        if (tid < off) red[tid] += red[tid + off];
        __syncthreads();
    }
    s = red[0];
    const float inv   = 1.0f / s;
    const float qs    = (float)(1 << qb[0]);
    const float invqs = 1.0f / qs;
    ushort_t* rp = route + (size_t)blockIdx.x * 8192;
#pragma unroll
    for (int i = 0; i < 32; ++i)
        rp[tid + i * 256] = f2bf(rintf(e[i] * inv * qs) * invqs);
}

// preactivate = sum_ci route*votes + bias; quant(qbr); squash over no; quant(qb)
__global__ __launch_bounds__(256) void route_weighted(const ushort_t* __restrict__ route,
                                                      const ushort_t* __restrict__ votes,
                                                      const float* __restrict__ bias,
                                                      const int* __restrict__ qbr,
                                                      const int* __restrict__ qb,
                                                      float* __restrict__ act) {
    int idx = blockIdx.x * 256 + threadIdx.x;    // (b,co,hw): 32*32*256
    int hw = idx & 255;
    int co = (idx >> 8) & 31;
    int b  = idx >> 13;
    float pre[8];
#pragma unroll
    for (int k = 0; k < 8; ++k) pre[k] = bias[co * 8 + k];
    const ushort_t* vbase = votes + ((size_t)b * CI * 256 + hw) * 256 + co * 8;
    for (int ci = 0; ci < CI; ++ci) {
        if (ci + 4 < CI)  // votes stride across ci = 64K halves; warm L0 from L2
            __builtin_prefetch(vbase + (size_t)(ci + 4) * 65536, 0, 0);
        float r = bf2f(route[(((size_t)b * CI + ci) * CO + co) * 256 + hw]);
        uint4 q = *(const uint4*)(vbase + (size_t)ci * 65536);
        float v[8];
        unpack8(q, v);
#pragma unroll
        for (int k = 0; k < 8; ++k) pre[k] += r * v[k];
    }
    const float qsr = (float)(1 << qbr[0]), invqsr = 1.0f / qsr;
    const float qsa = (float)(1 << qb[0]),  invqsa = 1.0f / qsa;
    float n2 = 0.0f;
#pragma unroll
    for (int k = 0; k < 8; ++k) {
        pre[k] = rintf(pre[k] * qsr) * invqsr;
        n2 += pre[k] * pre[k];
    }
    float scale = sqrtf(n2) / (1.0f + n2);       // squash: x*||x||/(1+||x||^2)
#pragma unroll
    for (int k = 0; k < 8; ++k) {
        float a = rintf(pre[k] * scale * qsa) * invqsa;
        act[(((size_t)b * CO + co) * NO + k) * 256 + hw] = a;
    }
}

// distances = sum_no act*votes; logits = quant(logits + distances, qbr)
__global__ __launch_bounds__(256) void agreement(const float* __restrict__ act,
                                                 const ushort_t* __restrict__ votes,
                                                 const int* __restrict__ qbr,
                                                 float* __restrict__ logits) {
    size_t idx = (size_t)blockIdx.x * 256 + threadIdx.x;  // [b][ci][co][hw] = 8388608
    int hw = (int)(idx & 255);
    int co = (int)((idx >> 8) & 31);
    int ci = (int)((idx >> 13) & 31);
    int b  = (int)(idx >> 18);
    const ushort_t* vp = votes + (((size_t)b * CI + ci) * 256 + hw) * 256 + co * 8;
    __builtin_prefetch(vp + 256, 0, 0);
    uint4 q = *(const uint4*)vp;
    float v[8];
    unpack8(q, v);
    float d = 0.0f;
#pragma unroll
    for (int k = 0; k < 8; ++k)
        d += act[(((size_t)b * CO + co) * NO + k) * 256 + hw] * v[k];
    const float qs = (float)(1 << qbr[0]), inv = 1.0f / qs;
    logits[idx] = rintf((logits[idx] + d) * qs) * inv;
}

// ---------------------------------------------------------------------------
extern "C" void kernel_launch(void* const* d_in, const int* in_sizes, int n_in,
                              void* d_out, int out_size, void* d_ws, size_t ws_size,
                              hipStream_t stream) {
    const float* x      = (const float*)d_in[0];
    const float* conv_w = (const float*)d_in[1];
    const float* conv_b = (const float*)d_in[2];
    const float* bias   = (const float*)d_in[3];
    const int*   qb     = (const int*)d_in[4];
    const int*   qbr    = (const int*)d_in[5];
    float* out = (float*)d_out;

    char* ws = (char*)d_ws;
    ushort_t* wfrag  = (ushort_t*)ws;                                   // 48 KB
    ushort_t* votes  = (ushort_t*)(ws + (64 << 10));                    // 134217728 B
    float*    logits = (float*)(ws + (64 << 10) + 134217728ull);        // 33554432 B
    ushort_t* route  = (ushort_t*)(ws + (64 << 10) + 134217728ull + 33554432ull); // 16 MB

    prep_wfrag<<<96, 256, 0, stream>>>(conv_w, wfrag);
    conv_votes<<<1024, 256, 0, stream>>>(x, wfrag, conv_b, qb, votes);
    zero_f32<<<32768, 256, 0, stream>>>(logits, 8388608);

    for (int it = 0; it < ITERATIONS; ++it) {
        softmax_route<<<1024, 256, 0, stream>>>(logits, route, qb);
        route_weighted<<<1024, 256, 0, stream>>>(route, votes, bias, qbr, qb, out);
        if (it + 1 < ITERATIONS)
            agreement<<<32768, 256, 0, stream>>>(out, votes, qbr, logits);
    }
}
